// DGCNN_2800318677499
// MI455X (gfx1250) — compile-verified, matrix operand order checked by reading, not run on hardware
//
#include <hip/hip_runtime.h>

typedef __attribute__((ext_vector_type(16))) __bf16 v16bf;
typedef __attribute__((ext_vector_type(8)))  __bf16 v8bf;
typedef __attribute__((ext_vector_type(8)))  float  v8f;
typedef __attribute__((ext_vector_type(4)))  float  v4f;

#define EPS 1e-5f
#define KNN 20
#define NPTS 2048
#define NB 8

#define CAT16(lo, hi) __builtin_shufflevector(lo, hi, 0, 1, 2, 3, 4, 5, 6, 7, \
                                              8, 9, 10, 11, 12, 13, 14, 15)

// ---------------------------------------------------------------------------
// [B,3,N] -> [B,N,3]
__global__ void transpose_kernel(const float* __restrict__ x, float* __restrict__ xt) {
  int e = blockIdx.x * blockDim.x + threadIdx.x;
  if (e >= NB * 3 * NPTS) return;
  int b = e / (3 * NPTS);
  int rem = e % (3 * NPTS);
  int c = rem / NPTS, n = rem % NPTS;
  xt[(size_t)(b * NPTS + n) * 3 + c] = x[e];
}

// f32 [O][K] -> bf16 [O][Kpad], zero padded K..Kpad
__global__ void tobf16_pad_kernel(const float* __restrict__ w, __bf16* __restrict__ o,
                                  int O, int K, int Kpad) {
  int e = blockIdx.x * blockDim.x + threadIdx.x;
  if (e >= O * Kpad) return;
  int r = e / Kpad, c = e % Kpad;
  o[e] = (c < K) ? (__bf16)w[r * K + c] : (__bf16)0.f;
}

// strided f32 features -> dense bf16 [BN][Cp], zero padded C..Cp
__global__ void featbf16_kernel(const float* __restrict__ X, int S, int off,
                                int C, int Cp, __bf16* __restrict__ out) {
  int e = blockIdx.x * blockDim.x + threadIdx.x;
  if (e >= NB * NPTS * Cp) return;
  int r = e / Cp, c = e % Cp;
  out[e] = (c < C) ? (__bf16)X[(size_t)r * S + off + c] : (__bf16)0.f;
}

// per-point squared norm
__global__ void sqnorm_kernel(const float* __restrict__ X, int S, int off, int C,
                              float* __restrict__ sq) {
  int r = blockIdx.x * blockDim.x + threadIdx.x;
  if (r >= NB * NPTS) return;
  const float* xr = X + (size_t)r * S + off;
  float s = 0.f;
  for (int c = 0; c < C; ++c) s += xr[c] * xr[c];
  sq[r] = s;
}

// ---------------------------------------------------------------------------
// WMMA Gram: score[b,n,m] = |x_m|^2 - 2 x_n.x_m (the row-constant |x_n|^2 is
// ordering-invariant for top-k and dropped). Block = 8 waves = 16 rows x 128
// cols; B-operand is the column tile's feature rows (same contiguous-K b128
// loads as a weight matrix).
__global__ void gram_kernel(const __bf16* __restrict__ Xbf, int Cp,
                            const float* __restrict__ sq, float* __restrict__ d2) {
  __shared__ alignas(16) __bf16 As[16 * 128];   // row-tile features (Cp<=128)
  const int tid = threadIdx.x, lane = tid & 31, wv = tid >> 5, half = lane >> 4;
  const int b  = blockIdx.x / (NPTS / 16);
  const int r0 = (blockIdx.x % (NPTS / 16)) * 16;
  const int m0 = blockIdx.y * 128 + wv * 16;
  for (int e = tid; e < 16 * Cp; e += 256)
    As[e] = Xbf[(size_t)(b * NPTS + r0 + e / Cp) * Cp + (e % Cp)];
  __syncthreads();
  const int colm = m0 + (lane & 15);
  const __bf16* Am = &As[(lane & 15) * Cp + 8 * half];
  const __bf16* Bm = Xbf + (size_t)(b * NPTS + colm) * Cp + 8 * half;
  v8f acc = {0.f, 0.f, 0.f, 0.f, 0.f, 0.f, 0.f, 0.f};
  for (int kc = 0; kc < Cp; kc += 32) {
    v8bf a0 = *(const v8bf*)(Am + kc);
    v8bf a1 = *(const v8bf*)(Am + kc + 16);
    v8bf b0 = *(const v8bf*)(Bm + kc);
    v8bf b1 = *(const v8bf*)(Bm + kc + 16);
    v16bf a  = CAT16(a0, a1);
    v16bf bb = CAT16(b0, b1);
    acc = __builtin_amdgcn_wmma_f32_16x16x32_bf16(false, a, false, bb,
                                                  (short)0, acc, false, false);
  }
  float sqm = sq[b * NPTS + colm];
  #pragma unroll
  for (int i = 0; i < 8; ++i) {
    int row = r0 + i + 8 * half;
    d2[(size_t)(b * NPTS + row) * NPTS + colm] = sqm - 2.f * acc[i];
  }
}

// ---------------------------------------------------------------------------
// Top-K selection (largest, low-index tie-break like jax.lax.top_k) from the
// precomputed score matrix. One wave per point row; 20 argmax extractions.
__global__ void topk_kernel(const float* __restrict__ d2, int* __restrict__ knn) {
  __shared__ alignas(16) float sc[4][NPTS];
  const int lane = threadIdx.x & 31;
  const int wv   = threadIdx.x >> 5;
  const int rowid = blockIdx.x * 4 + wv;          // b*N + n
  const float* row = d2 + (size_t)rowid * NPTS;
  for (int c0 = 0; c0 < NPTS / 4; c0 += 32) {
    int i4 = c0 + lane;
    v4f v = ((const v4f*)row)[i4];
    *(v4f*)&sc[wv][i4 * 4] = v;
  }
  __syncthreads();
  for (int it = 0; it < KNN; ++it) {
    float bv = -3.0e38f; int bi = NPTS;
    for (int m = lane; m < NPTS; m += 32) {
      float v = sc[wv][m];
      if (v > bv || (v == bv && m < bi)) { bv = v; bi = m; }
    }
    #pragma unroll
    for (int msk = 16; msk > 0; msk >>= 1) {
      float ov = __shfl_xor(bv, msk);
      int   oi = __shfl_xor(bi, msk);
      if (ov > bv || (ov == bv && oi < bi)) { bv = ov; bi = oi; }
    }
    if (lane == 0) { knn[(size_t)rowid * KNN + it] = bi; sc[wv][bi] = -3.0e38f; }
    __syncthreads();
  }
}

// ---------------------------------------------------------------------------
// Fused edge-conv: gather -> [nbr-ctr, ctr] -> bf16 WMMA GEMM -> running
// max/min over k (monotone-BN trick) + global channel sum/sumsq for BN stats.
__global__ void edgeconv_kernel(const float* __restrict__ X, int S, int off, int C,
                                int C2p, const __bf16* __restrict__ Wb, int O,
                                const int* __restrict__ knn,
                                float* __restrict__ hmax, float* __restrict__ hmin,
                                float* __restrict__ gsum, float* __restrict__ gsq) {
  __shared__ alignas(16) __bf16 F[16 * 256];  // 16 rows x C2p (<=256)
  __shared__ float  Xc[16 * 128];             // cached center rows (C<=128)
  __shared__ int    sidx[16 * KNN];
  const int tid  = threadIdx.x;
  const int lane = tid & 31;
  const int wv   = tid >> 5;
  const int half = lane >> 4;
  const int b    = blockIdx.x / (NPTS / 16);
  const int n0   = (blockIdx.x % (NPTS / 16)) * 16;
  const int C2   = 2 * C;
  const int O16  = O >> 4;

  for (int e = tid; e < 16 * C; e += 256) {
    int r = e / C, c = e % C;
    Xc[r * C + c] = X[(size_t)(b * NPTS + n0 + r) * S + off + c];
  }
  for (int e = tid; e < 16 * KNN; e += 256)
    sidx[e] = knn[(size_t)(b * NPTS + n0) * KNN + e];
  __syncthreads();

  __builtin_prefetch(Wb + (size_t)(wv * 16) * C2p, 0, 3);

  float rmax[2][8], rmin[2][8], ls[2], lq[2];
  #pragma unroll
  for (int oi = 0; oi < 2; ++oi) {
    ls[oi] = 0.f; lq[oi] = 0.f;
    #pragma unroll
    for (int i = 0; i < 8; ++i) { rmax[oi][i] = -3.0e38f; rmin[oi][i] = 3.0e38f; }
  }

  const __bf16* Fm = &F[(lane & 15) * C2p + 8 * half];

  for (int k = 0; k < KNN; ++k) {
    for (int e = tid; e < 16 * C2p; e += 256) {
      int r = e / C2p, c = e % C2p;
      float v = 0.f;
      if (c < C) {
        int nb = sidx[r * KNN + k];
        v = X[(size_t)(b * NPTS + nb) * S + off + c] - Xc[r * C + c];
      } else if (c < C2) {
        v = Xc[r * C + (c - C)];
      }
      F[e] = (__bf16)v;
    }
    __syncthreads();
    #pragma unroll
    for (int oi = 0; oi < 2; ++oi) {
      const int ot = wv + oi * 8;
      if (ot < O16) {
        const int col = ot * 16 + (lane & 15);
        const __bf16* Wr = Wb + (size_t)col * C2p + 8 * half;
        v8f acc = {0.f, 0.f, 0.f, 0.f, 0.f, 0.f, 0.f, 0.f};
        for (int kc = 0; kc < C2p; kc += 32) {
          v8bf a0 = *(const v8bf*)(Fm + kc);
          v8bf a1 = *(const v8bf*)(Fm + kc + 16);
          v8bf b0 = *(const v8bf*)(Wr + kc);
          v8bf b1 = *(const v8bf*)(Wr + kc + 16);
          v16bf a  = CAT16(a0, a1);
          v16bf bb = CAT16(b0, b1);
          acc = __builtin_amdgcn_wmma_f32_16x16x32_bf16(false, a, false, bb,
                                                        (short)0, acc, false, false);
        }
        #pragma unroll
        for (int i = 0; i < 8; ++i) {
          float h = acc[i];
          rmax[oi][i] = fmaxf(rmax[oi][i], h);
          rmin[oi][i] = fminf(rmin[oi][i], h);
          ls[oi] += h; lq[oi] += h * h;
        }
      }
    }
    __syncthreads();
  }
  #pragma unroll
  for (int oi = 0; oi < 2; ++oi) {
    const int ot = wv + oi * 8;
    if (ot < O16) {
      const int col = ot * 16 + (lane & 15);
      #pragma unroll
      for (int i = 0; i < 8; ++i) {
        int r = i + 8 * half;
        size_t e = (size_t)(b * NPTS + n0 + r) * O + col;
        hmax[e] = rmax[oi][i];
        hmin[e] = rmin[oi][i];
      }
      atomicAdd(&gsum[col], ls[oi]);
      atomicAdd(&gsq[col],  lq[oi]);
    }
  }
}

// BN(train) + leaky applied to max_k (or min_k when gamma<0) of h.
__global__ void bnmax_kernel(const float* __restrict__ hmax, const float* __restrict__ hmin,
                             const float* __restrict__ gsum, const float* __restrict__ gsq,
                             float cnt, const float* __restrict__ g, const float* __restrict__ bt,
                             float* __restrict__ out, int ldo, int oOff, int O, int total) {
  int e = blockIdx.x * blockDim.x + threadIdx.x;
  if (e >= total) return;
  int o = e % O, row = e / O;
  float m  = gsum[o] / cnt;
  float v  = gsq[o] / cnt - m * m;
  float rs = rsqrtf(v + EPS);
  float gg = g[o];
  float sel = (gg >= 0.f) ? hmax[e] : hmin[e];
  float y = gg * (sel - m) * rs + bt[o];
  out[(size_t)row * ldo + oOff + o] = (y > 0.f) ? y : 0.2f * y;
}

// ---------------------------------------------------------------------------
// z[16384,1024] = hc[16384,512] x w5^T, bf16 WMMA tiles (K=512).
__global__ void gemm_w5_kernel(const float* __restrict__ A, const __bf16* __restrict__ Wb,
                               float* __restrict__ Cm) {
  __shared__ alignas(16) __bf16 As[16 * 512];
  const int tid = threadIdx.x, lane = tid & 31, wv = tid >> 5, half = lane >> 4;
  const int m0 = blockIdx.x * 16;
  const int n0 = blockIdx.y * 128 + wv * 16;
  for (int e = tid; e < 16 * 512; e += 256)
    As[e] = (__bf16)A[(size_t)(m0 + (e >> 9)) * 512 + (e & 511)];
  __syncthreads();
  const int col = n0 + (lane & 15);
  const __bf16* Am = &As[(lane & 15) * 512 + 8 * half];
  const __bf16* Wr = Wb + (size_t)col * 512 + 8 * half;
  v8f acc = {0.f, 0.f, 0.f, 0.f, 0.f, 0.f, 0.f, 0.f};
  for (int kc = 0; kc < 512; kc += 32) {
    v8bf a0 = *(const v8bf*)(Am + kc);
    v8bf a1 = *(const v8bf*)(Am + kc + 16);
    v8bf b0 = *(const v8bf*)(Wr + kc);
    v8bf b1 = *(const v8bf*)(Wr + kc + 16);
    v16bf a  = CAT16(a0, a1);
    v16bf bb = CAT16(b0, b1);
    acc = __builtin_amdgcn_wmma_f32_16x16x32_bf16(false, a, false, bb,
                                                  (short)0, acc, false, false);
  }
  #pragma unroll
  for (int i = 0; i < 8; ++i)
    Cm[(size_t)(m0 + i + 8 * half) * 1024 + col] = acc[i];
}

// per-channel sum/sumsq over 16384 rows (one block per channel).
__global__ void chstats_kernel(const float* __restrict__ z, int M,
                               float* __restrict__ csum, float* __restrict__ csq) {
  __shared__ float ss[256], sq[256];
  int ch = blockIdx.x;
  float s = 0.f, q = 0.f;
  for (int r = threadIdx.x; r < M; r += 256) {
    float v = z[(size_t)r * 1024 + ch]; s += v; q += v * v;
  }
  ss[threadIdx.x] = s; sq[threadIdx.x] = q;
  __syncthreads();
  for (int st = 128; st; st >>= 1) {
    if (threadIdx.x < st) { ss[threadIdx.x] += ss[threadIdx.x + st];
                            sq[threadIdx.x] += sq[threadIdx.x + st]; }
    __syncthreads();
  }
  if (!threadIdx.x) { csum[ch] = ss[0]; csq[ch] = sq[0]; }
}

// BN + leaky + max/mean pool over N. One block per (b, channel).
__global__ void pool_kernel(const float* __restrict__ z, const float* __restrict__ csum,
                            const float* __restrict__ csq, const float* __restrict__ g,
                            const float* __restrict__ bt, float* __restrict__ pooled) {
  __shared__ float sm[256], ssum[256];
  int b = blockIdx.x >> 10;
  int ch = blockIdx.x & 1023;
  float m  = csum[ch] / 16384.f;
  float v  = csq[ch] / 16384.f - m * m;
  float rs = rsqrtf(v + EPS);
  float gg = g[ch], bb = bt[ch];
  float lmax = -3.0e38f, ls = 0.f;
  for (int n = threadIdx.x; n < NPTS; n += 256) {
    float x = z[(size_t)(b * NPTS + n) * 1024 + ch];
    float y = gg * (x - m) * rs + bb;
    y = (y > 0.f) ? y : 0.2f * y;
    lmax = fmaxf(lmax, y); ls += y;
  }
  sm[threadIdx.x] = lmax; ssum[threadIdx.x] = ls;
  __syncthreads();
  for (int st = 128; st; st >>= 1) {
    if (threadIdx.x < st) { sm[threadIdx.x] = fmaxf(sm[threadIdx.x], sm[threadIdx.x + st]);
                            ssum[threadIdx.x] += ssum[threadIdx.x + st]; }
    __syncthreads();
  }
  if (!threadIdx.x) {
    pooled[b * 2048 + ch]        = sm[0];
    pooled[b * 2048 + 1024 + ch] = ssum[0] / (float)NPTS;
  }
}

// small FC: out[b,o] = sum_c in[b,c]*W[o,c] (+bias)
__global__ void linear_kernel(const float* __restrict__ in, const float* __restrict__ W,
                              const float* __restrict__ bias, float* __restrict__ out,
                              int Bn, int Cin, int Cout) {
  int t = blockIdx.x * blockDim.x + threadIdx.x;
  if (t >= Bn * Cout) return;
  int b = t / Cout, o = t % Cout;
  float acc = bias ? bias[o] : 0.f;
  const float* ir = in + (size_t)b * Cin;
  const float* wr = W + (size_t)o * Cin;
  for (int c = 0; c < Cin; ++c) acc += ir[c] * wr[c];
  out[t] = acc;
}

// BN over batch (8) per channel + activation (0=leaky 0.2, 1=relu)
__global__ void bnbatch_kernel(const float* __restrict__ x, const float* __restrict__ g,
                               const float* __restrict__ bt, float* __restrict__ out,
                               int Cch, int ldo, int oOff, int relu_mode) {
  int o = blockIdx.x * blockDim.x + threadIdx.x;
  if (o >= Cch) return;
  float s = 0.f, q = 0.f;
  for (int b = 0; b < NB; ++b) { float v = x[b * Cch + o]; s += v; q += v * v; }
  float m = s / NB, var = q / NB - m * m, rs = rsqrtf(var + EPS);
  float gg = g[o], bb = bt[o];
  for (int b = 0; b < NB; ++b) {
    float y = gg * (x[b * Cch + o] - m) * rs + bb;
    y = relu_mode ? fmaxf(y, 0.f) : ((y > 0.f) ? y : 0.2f * y);
    out[b * ldo + oOff + o] = y;
  }
}

// ---------------------------------------------------------------------------
extern "C" void kernel_launch(void* const* d_in, const int* in_sizes, int n_in,
                              void* d_out, int out_size, void* d_ws, size_t ws_size,
                              hipStream_t stream) {
  (void)in_sizes; (void)n_in; (void)out_size; (void)ws_size;
  const float* xin[3] = {(const float*)d_in[0], (const float*)d_in[1], (const float*)d_in[2]};
  const float* w1 = (const float*)d_in[3];
  const float* w2 = (const float*)d_in[4];
  const float* w3 = (const float*)d_in[5];
  const float* w4 = (const float*)d_in[6];
  const float* w5 = (const float*)d_in[7];
  const float* wl1 = (const float*)d_in[8];
  const float* wl2 = (const float*)d_in[9];
  const float* bl2 = (const float*)d_in[10];
  const float* whead = (const float*)d_in[11];
  const float* bhead = (const float*)d_in[12];
  const float* g[8]; const float* be[8];
  for (int i = 0; i < 8; ++i) { g[i] = (const float*)d_in[13 + 2 * i];
                                be[i] = (const float*)d_in[14 + 2 * i]; }

  char* base = (char*)d_ws;
  size_t off = 0;
  auto alloc = [&](size_t bytes) {
    off = (off + 255) & ~(size_t)255; void* p = base + off; off += bytes; return p;
  };
  float* xt    = (float*)alloc((size_t)NB * NPTS * 3 * 4);
  float* hc    = (float*)alloc((size_t)NB * NPTS * 512 * 4);
  float* z     = (float*)alloc((size_t)NB * NPTS * 1024 * 4);
  float* hmax  = (float*)alloc((size_t)NB * NPTS * 256 * 4);
  float* hmin  = (float*)alloc((size_t)NB * NPTS * 256 * 4);
  float* d2    = (float*)alloc((size_t)NB * NPTS * NPTS * 4);   // 128 MB, L2-resident
  __bf16* xbf  = (__bf16*)alloc((size_t)NB * NPTS * 128 * 2);
  float* sqb   = (float*)alloc((size_t)NB * NPTS * 4);
  int*   knn   = (int*)alloc((size_t)NB * NPTS * KNN * 4);
  float* csum  = (float*)alloc(2048 * 4);
  float* csq   = csum + 1024;
  float* pooled = (float*)alloc(NB * 2048 * 4);
  float* lin1  = (float*)alloc(NB * 512 * 4);
  float* fc1   = (float*)alloc(NB * 512 * 4);
  float* lin2  = (float*)alloc(NB * 256 * 4);
  float* feats = (float*)alloc(NB * 768 * 4);
  float* hlin  = (float*)alloc(NB * 7 * 4);
  __bf16* wb1 = (__bf16*)alloc(64 * 32 * 2);      // padded 6 -> 32
  __bf16* wb2 = (__bf16*)alloc(64 * 128 * 2);
  __bf16* wb3 = (__bf16*)alloc(128 * 128 * 2);
  __bf16* wb4 = (__bf16*)alloc(256 * 256 * 2);
  __bf16* wb5 = (__bf16*)alloc(1024 * 512 * 2);

  auto cvt = [&](const float* s, __bf16* d, int O, int K, int Kp) {
    tobf16_pad_kernel<<<(O * Kp + 255) / 256, 256, 0, stream>>>(s, d, O, K, Kp);
  };
  cvt(w1, wb1, 64, 6, 32);
  cvt(w2, wb2, 64, 128, 128);
  cvt(w3, wb3, 128, 128, 128);
  cvt(w4, wb4, 256, 256, 256);
  cvt(w5, wb5, 1024, 512, 512);

  struct Layer { const float* X; int S, off, C, Cp, C2p; const __bf16* Wb; int O, outOff, gi; };

  for (int p = 0; p < 3; ++p) {
    transpose_kernel<<<(NB * 3 * NPTS + 255) / 256, 256, 0, stream>>>(xin[p], xt);
    Layer L[4] = {
      { xt, 3,   0,   3,   32,  32,  wb1, 64,  0,   0 },
      { hc, 512, 0,   64,  64,  128, wb2, 64,  64,  1 },
      { hc, 512, 64,  64,  64,  128, wb3, 128, 128, 2 },
      { hc, 512, 128, 128, 128, 256, wb4, 256, 256, 3 },
    };
    for (int l = 0; l < 4; ++l) {
      // --- kNN graph via WMMA Gram + LDS top-k extraction ---
      featbf16_kernel<<<(NB * NPTS * L[l].Cp + 255) / 256, 256, 0, stream>>>(
          L[l].X, L[l].S, L[l].off, L[l].C, L[l].Cp, xbf);
      sqnorm_kernel<<<(NB * NPTS + 255) / 256, 256, 0, stream>>>(
          L[l].X, L[l].S, L[l].off, L[l].C, sqb);
      gram_kernel<<<dim3(NB * NPTS / 16, NPTS / 128), 256, 0, stream>>>(
          xbf, L[l].Cp, sqb, d2);
      topk_kernel<<<NB * NPTS / 4, 128, 0, stream>>>(d2, knn);
      // --- fused edge-conv (WMMA) + BN/leaky/max ---
      (void)hipMemsetAsync(csum, 0, 2048 * 4, stream);
      edgeconv_kernel<<<NB * NPTS / 16, 256, 0, stream>>>(
          L[l].X, L[l].S, L[l].off, L[l].C, L[l].C2p, L[l].Wb, L[l].O,
          knn, hmax, hmin, csum, csq);
      int total = NB * NPTS * L[l].O;
      bnmax_kernel<<<(total + 255) / 256, 256, 0, stream>>>(
          hmax, hmin, csum, csq, (float)(NB * NPTS * KNN),
          g[L[l].gi], be[L[l].gi], hc, 512, L[l].outOff, L[l].O, total);
    }
    gemm_w5_kernel<<<dim3(NB * NPTS / 16, 8), 256, 0, stream>>>(hc, wb5, z);
    chstats_kernel<<<1024, 256, 0, stream>>>(z, NB * NPTS, csum, csq);
    pool_kernel<<<NB * 1024, 256, 0, stream>>>(z, csum, csq, g[4], be[4], pooled);
    linear_kernel<<<(NB * 512 + 255) / 256, 256, 0, stream>>>(pooled, wl1, nullptr, lin1, NB, 2048, 512);
    bnbatch_kernel<<<2, 256, 0, stream>>>(lin1, g[5], be[5], fc1, 512, 512, 0, 0);
    linear_kernel<<<(NB * 256 + 255) / 256, 256, 0, stream>>>(fc1, wl2, bl2, lin2, NB, 512, 256);
    bnbatch_kernel<<<1, 256, 0, stream>>>(lin2, g[6], be[6], feats, 256, 768, p * 256, 0);
  }
  linear_kernel<<<1, 64, 0, stream>>>(feats, whead, bhead, hlin, NB, 768, 7);
  bnbatch_kernel<<<1, 32, 0, stream>>>(hlin, g[7], be[7], (float*)d_out, 7, 7, 0, 1);
}
